// Attention_3332894621991
// MI455X (gfx1250) — compile-verified, matrix operand order checked by reading, not run on hardware
//
#include <hip/hip_runtime.h>

#define N_SEQ 4096
#define DIM 768
#define QKVN (3 * DIM)
#define HEADS 12
#define HDIM 64
#define SS 40  // LDS row stride (elements) for the 16x32 P tile; 80B rows keep 16B alignment

typedef __attribute__((ext_vector_type(16))) __bf16 bf16x16;
typedef __attribute__((ext_vector_type(8)))  __bf16 bf16x8;
typedef __attribute__((ext_vector_type(8)))  float  f32x8;
typedef __attribute__((ext_vector_type(4)))  int    i32x4;

#if defined(__gfx1250__) && __has_builtin(__builtin_amdgcn_global_load_async_to_lds_b128) && \
    __has_builtin(__builtin_amdgcn_s_wait_asynccnt)
#define USE_ASYNC_LDS 1
#define GPTR(p) ((i32x4*)(p))
#define LPTR(p) ((i32x4*)(p))
#else
#define USE_ASYNC_LDS 0
#endif

// ---------------------------------------------------------------------------
// WMMA wrapper: D(f32 16x16) = A(bf16 16x32) x B(bf16 32x16) + C
// ---------------------------------------------------------------------------
__device__ __forceinline__ f32x8 wmma_bf16(bf16x16 a, bf16x16 b, f32x8 c) {
  return __builtin_amdgcn_wmma_f32_16x16x32_bf16(false, a, false, b, (short)0, c,
                                                 false, false);
}

// Load one bf16 A/B fragment covering K in [koff, koff+32) for this lane.
// ISA layout (16-bit A 16x32): lane = h*16+m holds, in element e:
//   e in [0,8):  K = koff + 8*h + e          (one contiguous 16B chunk)
//   e in [8,16): K = koff + 16 + 8*h + (e-8) (second contiguous 16B chunk)
__device__ __forceinline__ bf16x16 load_frag(const __bf16* rowptr, int koff, int h) {
  bf16x8 lo = *(const bf16x8*)(rowptr + koff + 8 * h);
  bf16x8 hi = *(const bf16x8*)(rowptr + koff + 16 + 8 * h);
  bf16x16 r;
#pragma unroll
  for (int i = 0; i < 8; ++i) { r[i] = lo[i]; r[i + 8] = hi[i]; }
  return r;
}

// ---------------------------------------------------------------------------
// Prep kernels: fp32 -> bf16 conversion / transposition
// ---------------------------------------------------------------------------
__global__ void cvt_bf16(const float* __restrict__ in, __bf16* __restrict__ out, int n) {
  int i = blockIdx.x * blockDim.x + threadIdx.x;
  if (i < n) out[i] = (__bf16)in[i];
}

// W[K][C] (row-major) -> Wt[C][K] bf16
__global__ void transpose_bf16(const float* __restrict__ W, __bf16* __restrict__ Wt,
                               int K, int C) {
  int i = blockIdx.x * blockDim.x + threadIdx.x;
  if (i < K * C) {
    int c = i / K, k = i - c * K;
    Wt[i] = (__bf16)W[(size_t)k * C + c];
  }
}

// ---------------------------------------------------------------------------
// QKV GEMM: each wave computes a 16x64 tile (A fragment reused 4x, 4 acc chains)
// k-step issues A + all 4 B fragment loads before any WMMA so the waits stagger.
// Epilogue scatters into Q[h][n][64], K[h][n][64], Vt[h][64][n] (bf16).
// ---------------------------------------------------------------------------
__global__ __launch_bounds__(128) void qkv_gemm(const __bf16* __restrict__ X,
                                                const __bf16* __restrict__ Wt,
                                                __bf16* __restrict__ Qb,
                                                __bf16* __restrict__ Kb,
                                                __bf16* __restrict__ Vtb) {
  const int MT = N_SEQ / 16;          // 256 row tiles
  int wave = threadIdx.x >> 5, lane = threadIdx.x & 31;
  int tile = blockIdx.x * 4 + wave;   // 0 .. 256*36-1
  int tm = tile % MT, tn = tile / MT; // tn: 64-wide column tile, 0..35
  int m = lane & 15, h = lane >> 4;

  const __bf16* arow = X + (size_t)(tm * 16 + m) * DIM;
  const __bf16* brow[4];
#pragma unroll
  for (int b = 0; b < 4; ++b)
    brow[b] = Wt + (size_t)(tn * 64 + b * 16 + m) * DIM;

  f32x8 acc[4] = {};
#pragma unroll 2
  for (int k0 = 0; k0 < DIM; k0 += 32) {
    bf16x16 a = load_frag(arow, k0, h);
    bf16x16 bf[4];
#pragma unroll
    for (int b = 0; b < 4; ++b) bf[b] = load_frag(brow[b], k0, h);
#pragma unroll
    for (int b = 0; b < 4; ++b) acc[b] = wmma_bf16(a, bf[b], acc[b]);
  }

#pragma unroll
  for (int b = 0; b < 4; ++b) {
    int col  = tn * 64 + b * 16 + m;   // column this lane owns in C/D layout
    int seg  = col / DIM;              // 0=Q 1=K 2=V
    int cc   = col - seg * DIM;
    int head = cc / HDIM;
    int dd   = cc & (HDIM - 1);
#pragma unroll
    for (int v = 0; v < 8; ++v) {
      int r = tm * 16 + v + 8 * h;     // C/D layout: row = v + 8h
      __bf16 val = (__bf16)acc[b][v];
      if (seg == 0)      Qb[((size_t)head * N_SEQ + r) * HDIM + dd] = val;
      else if (seg == 1) Kb[((size_t)head * N_SEQ + r) * HDIM + dd] = val;
      else               Vtb[((size_t)head * HDIM + dd) * N_SEQ + r] = val;
    }
  }
}

// ---------------------------------------------------------------------------
// Flash attention core.
// Block = 4 waves on the SAME head, 64 consecutive queries (16 per wave).
// Per 32-key step: all 128 threads cooperatively stage the shared K tile
// (32x64) and V tile (64x32) into LDS via async global->LDS (ASYNCcnt path),
// then each wave runs 4 S-WMMAs, online softmax, P->LDS relayout, 4 PV-WMMAs.
// ---------------------------------------------------------------------------
__global__ __launch_bounds__(128) void attn_core(const __bf16* __restrict__ Q,
                                                 const __bf16* __restrict__ Kk,
                                                 const __bf16* __restrict__ Vt,
                                                 __bf16* __restrict__ Aout) {
  __shared__ __bf16 Ksh[32 * HDIM];      // [key row 0..31][d 0..63]
  __shared__ __bf16 Vsh[HDIM * 32];      // [d col 0..63][key 0..31]
  __shared__ __bf16 Psh[4][16 * SS];     // per-wave P relayout scratch

  int tid  = threadIdx.x;
  int wave = tid >> 5, lane = tid & 31;
  int head = blockIdx.y;
  int qb   = blockIdx.x * 64 + wave * 16;
  int m = lane & 15, h = lane >> 4;

  const __bf16* Qh = Q  + (size_t)head * N_SEQ * HDIM;
  const __bf16* Kh = Kk + (size_t)head * N_SEQ * HDIM;
  const __bf16* Vh = Vt + (size_t)head * HDIM * N_SEQ;
  __bf16* psh = Psh[wave];

  // Q fragments: 16x64 split into two K=32 fragments, held for the whole loop
  const __bf16* qrow = Qh + (size_t)(qb + m) * HDIM;
  bf16x16 qa0 = load_frag(qrow, 0, h);
  bf16x16 qa1 = load_frag(qrow, 32, h);

  float mr[8], lr[8];
  f32x8 o[4] = {};
#pragma unroll
  for (int v = 0; v < 8; ++v) { mr[v] = -3.0e38f; lr[v] = 0.0f; }

  for (int j0 = 0; j0 < N_SEQ; j0 += 32) {
    // ---- cooperative staging: 256 16B chunks each for K and V ----
#pragma unroll
    for (int i = 0; i < 2; ++i) {
      int id = tid * 2 + i;                                    // 0..255
      int kr = id >> 3, kc = id & 7;                           // K: row, 8-elem chunk
      int vr = id >> 2, vc = id & 3;                           // V: d-col, 8-elem chunk
      const __bf16* gk = Kh + (size_t)(j0 + kr) * HDIM + kc * 8;
      const __bf16* gv = Vh + (size_t)vr * N_SEQ + j0 + vc * 8;
#if USE_ASYNC_LDS
      __builtin_amdgcn_global_load_async_to_lds_b128(GPTR(gk), LPTR(&Ksh[id * 8]), 0, 0);
      __builtin_amdgcn_global_load_async_to_lds_b128(GPTR(gv), LPTR(&Vsh[id * 8]), 0, 0);
#else
      *(bf16x8*)&Ksh[id * 8] = *(const bf16x8*)gk;
      *(bf16x8*)&Vsh[id * 8] = *(const bf16x8*)gv;
#endif
    }
#if USE_ASYNC_LDS
    __builtin_amdgcn_s_wait_asynccnt(0);
#endif
    __syncthreads();

    // ---- S = Q K^T for 32 keys (two 16-key groups), B fragments from LDS ----
    f32x8 s0 = {}, s1 = {};
    s0 = wmma_bf16(qa0, load_frag(Ksh + m * HDIM, 0, h), s0);
    s0 = wmma_bf16(qa1, load_frag(Ksh + m * HDIM, 32, h), s0);
    s1 = wmma_bf16(qa0, load_frag(Ksh + (16 + m) * HDIM, 0, h), s1);
    s1 = wmma_bf16(qa1, load_frag(Ksh + (16 + m) * HDIM, 32, h), s1);

    // ---- online softmax; D layout row = v + 8h, cols across 16-lane half ----
#pragma unroll
    for (int v = 0; v < 8; ++v) {
      float a0 = s0[v], a1 = s1[v];
      float tm2 = fmaxf(a0, a1);
      tm2 = fmaxf(tm2, __shfl_xor(tm2, 1, 32));
      tm2 = fmaxf(tm2, __shfl_xor(tm2, 2, 32));
      tm2 = fmaxf(tm2, __shfl_xor(tm2, 4, 32));
      tm2 = fmaxf(tm2, __shfl_xor(tm2, 8, 32));
      float mn = fmaxf(mr[v], tm2);
      float sc = __expf(mr[v] - mn);
      float p0 = __expf(a0 - mn);
      float p1 = __expf(a1 - mn);
      float rs = p0 + p1;
      rs += __shfl_xor(rs, 1, 32);
      rs += __shfl_xor(rs, 2, 32);
      rs += __shfl_xor(rs, 4, 32);
      rs += __shfl_xor(rs, 8, 32);
      lr[v] = lr[v] * sc + rs;
      mr[v] = mn;
      o[0][v] *= sc; o[1][v] *= sc; o[2][v] *= sc; o[3][v] *= sc;
      int r = v + 8 * h;
      psh[r * SS + m]      = (__bf16)p0;   // local key cols 0..15
      psh[r * SS + 16 + m] = (__bf16)p1;   // local key cols 16..31
    }

    // ---- P (16x32) back as A-fragment; same-wave LDS is in-order ----
    bf16x16 pa = load_frag(psh + m * SS, 0, h);

    // ---- O += P @ V from the shared V tile ----
    bf16x16 vf[4];
#pragma unroll
    for (int dg = 0; dg < 4; ++dg) vf[dg] = load_frag(Vsh + (dg * 16 + m) * 32, 0, h);
#pragma unroll
    for (int dg = 0; dg < 4; ++dg) o[dg] = wmma_bf16(pa, vf[dg], o[dg]);

    __syncthreads();  // protect K/V tiles before next staging round
  }

  // ---- normalize and emit bf16 [n][h*d] for the out-projection GEMM ----
#pragma unroll
  for (int v = 0; v < 8; ++v) {
    float inv = 1.0f / lr[v];
    size_t base = (size_t)(qb + v + 8 * h) * DIM + head * HDIM + m;
    Aout[base +  0] = (__bf16)(o[0][v] * inv);
    Aout[base + 16] = (__bf16)(o[1][v] * inv);
    Aout[base + 32] = (__bf16)(o[2][v] * inv);
    Aout[base + 48] = (__bf16)(o[3][v] * inv);
  }
}

// ---------------------------------------------------------------------------
// Output projection: each wave computes a 16x64 tile; fp32 out + bias
// ---------------------------------------------------------------------------
__global__ __launch_bounds__(128) void out_gemm(const __bf16* __restrict__ A,
                                                const __bf16* __restrict__ Wt,
                                                const float* __restrict__ bias,
                                                float* __restrict__ out) {
  const int MT = N_SEQ / 16;          // 256
  int wave = threadIdx.x >> 5, lane = threadIdx.x & 31;
  int tile = blockIdx.x * 4 + wave;   // 0 .. 256*12-1
  int tm = tile % MT, tn = tile / MT; // tn: 64-wide column tile, 0..11
  int m = lane & 15, h = lane >> 4;

  const __bf16* arow = A + (size_t)(tm * 16 + m) * DIM;
  const __bf16* brow[4];
#pragma unroll
  for (int b = 0; b < 4; ++b)
    brow[b] = Wt + (size_t)(tn * 64 + b * 16 + m) * DIM;

  f32x8 acc[4] = {};
#pragma unroll 2
  for (int k0 = 0; k0 < DIM; k0 += 32) {
    bf16x16 a = load_frag(arow, k0, h);
    bf16x16 bf[4];
#pragma unroll
    for (int b = 0; b < 4; ++b) bf[b] = load_frag(brow[b], k0, h);
#pragma unroll
    for (int b = 0; b < 4; ++b) acc[b] = wmma_bf16(a, bf[b], acc[b]);
  }

#pragma unroll
  for (int b = 0; b < 4; ++b) {
    int col = tn * 64 + b * 16 + m;
    float bv = bias[col];
#pragma unroll
    for (int v = 0; v < 8; ++v)
      out[(size_t)(tm * 16 + v + 8 * h) * DIM + col] = acc[b][v] + bv;
  }
}

// ---------------------------------------------------------------------------
// Host-side launch
// ---------------------------------------------------------------------------
extern "C" void kernel_launch(void* const* d_in, const int* in_sizes, int n_in,
                              void* d_out, int out_size, void* d_ws, size_t ws_size,
                              hipStream_t stream) {
  const float* x     = (const float*)d_in[0];   // [1,4096,768]
  const float* W_qkv = (const float*)d_in[1];   // [768,2304]
  const float* W_out = (const float*)d_in[2];   // [768,768]
  const float* b_out = (const float*)d_in[3];   // [768]
  float* out = (float*)d_out;

  char* ws = (char*)d_ws;
  size_t off = 0;
  auto alloc = [&](size_t bytes) { char* p = ws + off; off += (bytes + 255) & ~(size_t)255; return p; };
  __bf16* Xbf   = (__bf16*)alloc((size_t)N_SEQ * DIM * 2);
  __bf16* Wqkvt = (__bf16*)alloc((size_t)QKVN * DIM * 2);
  __bf16* Woutt = (__bf16*)alloc((size_t)DIM * DIM * 2);
  __bf16* Qb    = (__bf16*)alloc((size_t)HEADS * N_SEQ * HDIM * 2);
  __bf16* Kb    = (__bf16*)alloc((size_t)HEADS * N_SEQ * HDIM * 2);
  __bf16* Vtb   = (__bf16*)alloc((size_t)HEADS * HDIM * N_SEQ * 2);
  __bf16* Abf   = (__bf16*)alloc((size_t)N_SEQ * DIM * 2);
  (void)ws_size; (void)n_in; (void)in_sizes; (void)out_size;

  // 1) precision prep
  {
    int n = N_SEQ * DIM;
    cvt_bf16<<<(n + 255) / 256, 256, 0, stream>>>(x, Xbf, n);
  }
  transpose_bf16<<<(QKVN * DIM + 255) / 256, 256, 0, stream>>>(W_qkv, Wqkvt, DIM, QKVN);
  transpose_bf16<<<(DIM * DIM + 255) / 256, 256, 0, stream>>>(W_out, Woutt, DIM, DIM);

  // 2) QKV projection + head split: 256 x 36 waves of 16x64 tiles
  qkv_gemm<<<(256 * (QKVN / 64)) / 4, 128, 0, stream>>>(Xbf, Wqkvt, Qb, Kb, Vtb);

  // 3) attention core: grid (query 64-blocks, heads)
  {
    dim3 grid(N_SEQ / 64, HEADS);
    attn_core<<<grid, 128, 0, stream>>>(Qb, Kb, Vtb, Abf);
  }

  // 4) output projection: 256 x 12 waves of 16x64 tiles
  out_gemm<<<(256 * (DIM / 64)) / 4, 128, 0, stream>>>(Abf, Woutt, b_out, out);
}
